// SDPA_31310311588130
// MI455X (gfx1250) — matrix operand order repaired and checked by reading, not compiled, not run on hardware
//
#include <hip/hip_runtime.h>
#include <hip/hip_bf16.h>

typedef __attribute__((ext_vector_type(16))) _Float16 v16h;
typedef __attribute__((ext_vector_type(8)))  _Float16 v8h;
typedef __attribute__((ext_vector_type(8)))  float    v8f;
typedef __attribute__((ext_vector_type(4)))  float    v4f;

#define HW_DIM  4096
#define C_DIM   64
#define NBATCH  4
#define LDS_STRIDE 72   // halves; 144B rows: 16B-aligned b128, conflict-friendly

static constexpr float kScale  = 0.125f;   // 1/sqrt(64)
static constexpr float kNegBig = -1.0e30f;

// 8 floats -> 8 halves
static __device__ inline v8h cvt8(v4f a, v4f b) {
  v8h h;
  h[0] = (_Float16)a.x; h[1] = (_Float16)a.y; h[2] = (_Float16)a.z; h[3] = (_Float16)a.w;
  h[4] = (_Float16)b.x; h[5] = (_Float16)b.y; h[6] = (_Float16)b.z; h[7] = (_Float16)b.w;
  return h;
}

// A-fragment chunk (ISA 7.12.2 layout) from an f16 LDS row:
// halves 0..7 at row[off], halves 8..15 at row[off+16]; off = chunkBase + hi*8.
static __device__ inline v16h lds_a_chunk(const _Float16* row, int off) {
  const v8h lo = *(const v8h*)(row + off);
  const v8h hh = *(const v8h*)(row + off + 16);
  return __builtin_shufflevector(lo, hh, 0,1,2,3,4,5,6,7,8,9,10,11,12,13,14,15);
}

// 16 consecutive halves from an f16 LDS row (B-fragment row slice).
static __device__ inline v16h lds_row16(const _Float16* p) {
  const v8h lo = *(const v8h*)(p);
  const v8h hh = *(const v8h*)(p + 8);
  return __builtin_shufflevector(lo, hh, 0,1,2,3,4,5,6,7,8,9,10,11,12,13,14,15);
}

static __device__ inline v8f wmma_f16(v16h a, v16h b, v8f c) {
  return __builtin_amdgcn_wmma_f32_16x16x32_f16(false, a, false, b, (short)0, c, false, false);
}

// ---------------------------------------------------------------------------
// Pass 1: column softmax stats m[k], 1/s[k] (softmax over q per column k).
// Block = 4 waves; each wave owns 64 k columns. The 16x64 Q tile is staged in
// LDS (f16, double-buffered) once per block and shared by all 4 waves.
// ---------------------------------------------------------------------------
#define P1_KT 4

__global__ __launch_bounds__(128) void sdpa_colstats(
    const float* __restrict__ Q, const float* __restrict__ K,
    float* __restrict__ Mcol, float* __restrict__ Rcol)
{
  __shared__ __align__(16) _Float16 qs[2][16 * LDS_STRIDE];

  const int tid  = threadIdx.x;
  const int lane = tid & 31;
  const int wave = tid >> 5;
  const int hi   = lane >> 4;
  const int mrow = lane & 15;
  const int n    = blockIdx.y;
  const int kb   = (blockIdx.x * 4 + wave) * (P1_KT * 16);

  const float* __restrict__ Qn = Q + (size_t)n * HW_DIM * C_DIM;
  const float* __restrict__ Kn = K + (size_t)n * HW_DIM * C_DIM;

  // B fragments of K^T (per wave, loaded once): lane%16 = kcol, half j = c in table order.
  v16h bk[P1_KT][2];
  #pragma unroll
  for (int t = 0; t < P1_KT; ++t) {
    #pragma unroll
    for (int ch = 0; ch < 2; ++ch) {
      const float* col = Kn + (size_t)(kb + t * 16 + mrow) * C_DIM + (ch * 32 + hi * 8);
      v16h tmp;
      #pragma unroll
      for (int j = 0; j < 16; ++j) tmp[j] = (_Float16)col[j + (j >= 8 ? 8 : 0)];
      bk[t][ch] = tmp;
    }
  }

  // Cooperative staging coords: 128 threads cover 16 rows x 64 floats.
  const int srow = tid >> 3;         // 0..15
  const int scol = (tid & 7) * 8;    // 0..56

  {  // prologue: stage q-tile 0
    const float* g = Qn + (size_t)srow * C_DIM + scol;
    *(v8h*)&qs[0][srow * LDS_STRIDE + scol] = cvt8(*(const v4f*)g, *(const v4f*)(g + 4));
  }

  float mrun[P1_KT], srun[P1_KT];
  #pragma unroll
  for (int t = 0; t < P1_KT; ++t) { mrun[t] = kNegBig; srun[t] = 0.0f; }

  const int niter = HW_DIM / 16;
  for (int it = 0; it < niter; ++it) {
    __syncthreads();
    const int buf = it & 1;
    if (it + 1 < niter) {  // stage next tile into the other buffer
      const float* g = Qn + (size_t)((it + 1) * 16 + srow) * C_DIM + scol;
      *(v8h*)&qs[buf ^ 1][srow * LDS_STRIDE + scol] = cvt8(*(const v4f*)g, *(const v4f*)(g + 4));
      if (it + 2 < niter)
        __builtin_prefetch(Qn + (size_t)((it + 2) * 16 + srow) * C_DIM + scol, 0, 0);
    }
    const _Float16* row = &qs[buf][mrow * LDS_STRIDE];
    const v16h a0 = lds_a_chunk(row, hi * 8);
    const v16h a1 = lds_a_chunk(row, 32 + hi * 8);

    #pragma unroll
    for (int t = 0; t < P1_KT; ++t) {
      v8f acc = {};
      acc = wmma_f16(a0, bk[t][0], acc);
      acc = wmma_f16(a1, bk[t][1], acc);
      // lane holds one column k across 8 q rows -> in-lane online softmax update
      float tmax = kNegBig;
      #pragma unroll
      for (int j = 0; j < 8; ++j) tmax = fmaxf(tmax, acc[j]);
      tmax *= kScale;
      const float nm = fmaxf(mrun[t], tmax);
      float ssum = 0.0f;
      #pragma unroll
      for (int j = 0; j < 8; ++j) ssum += __expf(acc[j] * kScale - nm);
      srun[t] = srun[t] * __expf(mrun[t] - nm) + ssum;
      mrun[t] = nm;
    }
  }

  // Merge the two half-wave partials for the same column (lane <-> lane^16).
  #pragma unroll
  for (int t = 0; t < P1_KT; ++t) {
    const float om = __shfl_xor(mrun[t], 16, 32);
    const float os = __shfl_xor(srun[t], 16, 32);
    const float nm = fmaxf(mrun[t], om);
    const float s  = srun[t] * __expf(mrun[t] - nm) + os * __expf(om - nm);
    if (lane < 16) {
      const int k = kb + t * 16 + mrow;
      Mcol[n * HW_DIM + k] = nm;
      Rcol[n * HW_DIM + k] = 1.0f / s;
    }
  }
}

// ---------------------------------------------------------------------------
// Pass 2: Out[q,c] = sum_k exp(l[q,k]*scale - m[k]) * r[k] * V[k,c].
// Logits computed TRANSPOSED (S^T = K*Q^T) so S^T C-fragments feed the PV
// GEMM's A-fragments with zero cross-lane movement. Block = 4 waves, each wave
// owns 32 q rows; the 32x64 K and V chunks are staged in LDS (f16, double-
// buffered) once per block and shared by all 4 waves.
// ---------------------------------------------------------------------------
#define P2_QT 2

__global__ __launch_bounds__(128) void sdpa_attn_out(
    const float* __restrict__ Q, const float* __restrict__ K,
    const float* __restrict__ V,
    const float* __restrict__ Mcol, const float* __restrict__ Rcol,
    float* __restrict__ Out)
{
  __shared__ __align__(16) _Float16 ks[2][32 * LDS_STRIDE];
  __shared__ __align__(16) _Float16 vs[2][32 * LDS_STRIDE];

  const int tid  = threadIdx.x;
  const int lane = tid & 31;
  const int wave = tid >> 5;
  const int hi   = lane >> 4;
  const int mrow = lane & 15;
  const int n    = blockIdx.y;
  const int qb   = (blockIdx.x * 4 + wave) * (P2_QT * 16);

  const float* __restrict__ Qn = Q + (size_t)n * HW_DIM * C_DIM;
  const float* __restrict__ Kn = K + (size_t)n * HW_DIM * C_DIM;
  const float* __restrict__ Vn = V + (size_t)n * HW_DIM * C_DIM;
  const float* __restrict__ Mn = Mcol + n * HW_DIM;
  const float* __restrict__ Rn = Rcol + n * HW_DIM;

  // B fragments of Q^T (per wave, loaded once): B[c][qcol] = Q[qcol][c]; lane%16 = qcol, half j = c in table order.
  v16h bq[P2_QT][2];
  #pragma unroll
  for (int qt = 0; qt < P2_QT; ++qt) {
    #pragma unroll
    for (int ch = 0; ch < 2; ++ch) {
      const float* col = Qn + (size_t)(qb + qt * 16 + mrow) * C_DIM + (ch * 32 + hi * 8);
      v16h tmp;
      #pragma unroll
      for (int j = 0; j < 16; ++j) tmp[j] = (_Float16)col[j + (j >= 8 ? 8 : 0)];
      bq[qt][ch] = tmp;
    }
  }

  v8f acc[P2_QT][4];
  #pragma unroll
  for (int qt = 0; qt < P2_QT; ++qt)
    #pragma unroll
    for (int ct = 0; ct < 4; ++ct) acc[qt][ct] = (v8f){};

  // Cooperative staging coords: 128 threads cover 32 rows x 64 floats per array.
  const int srow = tid >> 2;         // 0..31
  const int scol = (tid & 3) * 16;   // 0,16,32,48

  {  // prologue: stage k-chunk 0
    const float* gk = Kn + (size_t)srow * C_DIM + scol;
    const float* gv = Vn + (size_t)srow * C_DIM + scol;
    _Float16* dk = &ks[0][srow * LDS_STRIDE + scol];
    _Float16* dv = &vs[0][srow * LDS_STRIDE + scol];
    *(v8h*)(dk)     = cvt8(*(const v4f*)(gk),     *(const v4f*)(gk + 4));
    *(v8h*)(dk + 8) = cvt8(*(const v4f*)(gk + 8), *(const v4f*)(gk + 12));
    *(v8h*)(dv)     = cvt8(*(const v4f*)(gv),     *(const v4f*)(gv + 4));
    *(v8h*)(dv + 8) = cvt8(*(const v4f*)(gv + 8), *(const v4f*)(gv + 12));
  }

  const int niter = HW_DIM / 32;
  for (int it = 0; it < niter; ++it) {
    __syncthreads();
    const int buf = it & 1;
    const int k0  = it * 32;

    if (it + 1 < niter) {  // stage next K/V chunk into the other buffer
      const float* gk = Kn + (size_t)((it + 1) * 32 + srow) * C_DIM + scol;
      const float* gv = Vn + (size_t)((it + 1) * 32 + srow) * C_DIM + scol;
      _Float16* dk = &ks[buf ^ 1][srow * LDS_STRIDE + scol];
      _Float16* dv = &vs[buf ^ 1][srow * LDS_STRIDE + scol];
      *(v8h*)(dk)     = cvt8(*(const v4f*)(gk),     *(const v4f*)(gk + 4));
      *(v8h*)(dk + 8) = cvt8(*(const v4f*)(gk + 8), *(const v4f*)(gk + 12));
      *(v8h*)(dv)     = cvt8(*(const v4f*)(gv),     *(const v4f*)(gv + 4));
      *(v8h*)(dv + 8) = cvt8(*(const v4f*)(gv + 8), *(const v4f*)(gv + 12));
      if (it + 2 < niter) {
        __builtin_prefetch(Kn + (size_t)((it + 2) * 32 + srow) * C_DIM + scol, 0, 0);
        __builtin_prefetch(Vn + (size_t)((it + 2) * 32 + srow) * C_DIM + scol, 0, 0);
      }
    }

    // A-fragments of the two 16-row K tiles (M = k, Kdim = c), from LDS.
    v16h aK[2][2];
    #pragma unroll
    for (int kt = 0; kt < 2; ++kt) {
      const _Float16* row = &ks[buf][(kt * 16 + mrow) * LDS_STRIDE];
      aK[kt][0] = lds_a_chunk(row, hi * 8);
      aK[kt][1] = lds_a_chunk(row, 32 + hi * 8);
    }
    // B fragments of the V chunk (32 k-rows x 16 c-cols): lane%16 = c col, half j = k row in table order.
    v16h bv[4];
    {
      const _Float16* vcol = &vs[buf][hi * 8 * LDS_STRIDE + mrow];
      #pragma unroll
      for (int ct = 0; ct < 4; ++ct)
        #pragma unroll
        for (int j = 0; j < 16; ++j) bv[ct][j] = vcol[(j + (j >= 8 ? 8 : 0)) * LDS_STRIDE + ct * 16];
    }
    // Column stats for these 32 k's, per C-layout rows (vgpr j -> k = base + j).
    v4f mk[2][2], rk[2][2];
    #pragma unroll
    for (int kt = 0; kt < 2; ++kt) {
      const int base = k0 + kt * 16 + hi * 8;
      mk[kt][0] = *(const v4f*)(Mn + base);
      mk[kt][1] = *(const v4f*)(Mn + base + 4);
      rk[kt][0] = *(const v4f*)(Rn + base);
      rk[kt][1] = *(const v4f*)(Rn + base + 4);
    }

    #pragma unroll
    for (int qt = 0; qt < P2_QT; ++qt) {
      v8f t0 = {}, t1 = {};
      t0 = wmma_f16(aK[0][0], bq[qt][0], t0);
      t0 = wmma_f16(aK[0][1], bq[qt][1], t0);
      t1 = wmma_f16(aK[1][0], bq[qt][0], t1);
      t1 = wmma_f16(aK[1][1], bq[qt][1], t1);
      // S^T C-fragments map directly onto the PV A-fragment: halves 0..7 <- t0, 8..15 <- t1.
      v16h a2;
      #pragma unroll
      for (int j = 0; j < 4; ++j) {
        a2[j]      = (_Float16)(__expf(t0[j]     * kScale - mk[0][0][j]) * rk[0][0][j]);
        a2[4 + j]  = (_Float16)(__expf(t0[4 + j] * kScale - mk[0][1][j]) * rk[0][1][j]);
        a2[8 + j]  = (_Float16)(__expf(t1[j]     * kScale - mk[1][0][j]) * rk[1][0][j]);
        a2[12 + j] = (_Float16)(__expf(t1[4 + j] * kScale - mk[1][1][j]) * rk[1][1][j]);
      }
      #pragma unroll
      for (int ct = 0; ct < 4; ++ct)
        acc[qt][ct] = wmma_f16(a2, bv[ct], acc[qt][ct]);
    }
  }

  // Store: C-layout lane -> c = ct*16 + mrow, vgpr j -> q = qb + qt*16 + hi*8 + j.
  #pragma unroll
  for (int qt = 0; qt < P2_QT; ++qt) {
    #pragma unroll
    for (int ct = 0; ct < 4; ++ct) {
      #pragma unroll
      for (int j = 0; j < 8; ++j) {
        const size_t qi = (size_t)n * HW_DIM + qb + qt * 16 + hi * 8 + j;
        Out[qi * C_DIM + ct * 16 + mrow] = acc[qt][ct][j];
      }
    }
  }
}

extern "C" void kernel_launch(void* const* d_in, const int* in_sizes, int n_in,
                              void* d_out, int out_size, void* d_ws, size_t ws_size,
                              hipStream_t stream) {
  (void)in_sizes; (void)n_in; (void)out_size; (void)ws_size;
  const float* Q = (const float*)d_in[0];
  const float* K = (const float*)d_in[1];
  const float* V = (const float*)d_in[2];
  float* Out  = (float*)d_out;
  float* Mcol = (float*)d_ws;                       // NBATCH*HW floats
  float* Rcol = Mcol + (size_t)NBATCH * HW_DIM;     // NBATCH*HW floats

  dim3 b(128);
  dim3 g1(HW_DIM / (4 * P1_KT * 16), NBATCH);       // 16 x 4
  sdpa_colstats<<<g1, b, 0, stream>>>(Q, K, Mcol, Rcol);

  dim3 g2(HW_DIM / (4 * P2_QT * 16), NBATCH);       // 32 x 4
  sdpa_attn_out<<<g2, b, 0, stream>>>(Q, K, V, Mcol, Rcol, Out);
}
